// CrossAttention_24575802868332
// MI455X (gfx1250) — matrix operand order repaired and checked by reading, not compiled, not run on hardware
//
#include <hip/hip_runtime.h>

#ifndef __has_builtin
#define __has_builtin(x) 0
#endif

typedef __bf16 bf16_t;
typedef __attribute__((ext_vector_type(16))) __bf16 v16bf;
typedef __attribute__((ext_vector_type(8)))  __bf16 v8bf;
typedef __attribute__((ext_vector_type(4)))  __bf16 v4bf;
typedef __attribute__((ext_vector_type(8)))  float  v8f;
typedef __attribute__((ext_vector_type(4)))  int    v4i;

#define GLOBAL_AS __attribute__((address_space(1)))
#define LDS_AS    __attribute__((address_space(3)))

// round-to-nearest-even f32 -> bf16 (pure bit ops; no __bf16 arithmetic needed)
static __device__ __forceinline__ bf16_t f2bf(float f) {
    unsigned u = __builtin_bit_cast(unsigned, f);
    u += 0x7FFFu + ((u >> 16) & 1u);
    unsigned short h = (unsigned short)(u >> 16);
    return __builtin_bit_cast(bf16_t, h);
}

// build a 16-wide bf16 fragment from two 16-byte chunks
static __device__ __forceinline__ v16bf mk16(const bf16_t* p0, const bf16_t* p1) {
    v8bf lo = *(const v8bf*)p0;
    v8bf hi = *(const v8bf*)p1;
    return __builtin_shufflevector(lo, hi, 0,1,2,3,4,5,6,7,8,9,10,11,12,13,14,15);
}

static __device__ __forceinline__ v8f wmma_bf16(v16bf a, v16bf b, v8f c) {
    return __builtin_amdgcn_wmma_f32_16x16x32_bf16(false, a, false, b, (short)0, c, false, false);
}

// 16-byte global -> LDS copy; async (ASYNCcnt-tracked, no VGPR round trip) when
// the gfx1250 builtin is available, plain copy otherwise.
// Builtin signature (from hipcc diagnostic): (v4i AS1*, v4i AS3*, imm offset, imm cpol)
static __device__ __forceinline__ void async_copy16(bf16_t* lds_dst, const bf16_t* gsrc) {
#if __has_builtin(__builtin_amdgcn_global_load_async_to_lds_b128)
    __builtin_amdgcn_global_load_async_to_lds_b128(
        (GLOBAL_AS v4i*)(void*)gsrc, (LDS_AS v4i*)(void*)lds_dst, 0, 0);
#else
    *(v8bf*)lds_dst = *(const v8bf*)gsrc;
#endif
}

static __device__ __forceinline__ void wait_async() {
#if __has_builtin(__builtin_amdgcn_s_wait_asynccnt)
    __builtin_amdgcn_s_wait_asynccnt(0);
#endif
}

static __device__ __forceinline__ void stage4(bf16_t* d, const float* s) {
    float4 t = *(const float4*)s;
    v4bf v; v[0] = f2bf(t.x); v[1] = f2bf(t.y); v[2] = f2bf(t.z); v[3] = f2bf(t.w);
    *(v4bf*)d = v;
}

// ---------------------------------------------------------------------------
// Generic WMMA GEMM: C[M,N] = A[M,K] * B[K,N]
// Block: 256 threads (8 waves), 128x64 output tile, K-step 32.
// MODE 0: out bf16, scaled by alpha (Q/K projections)
// MODE 1: out bf16, scattered to Vt[b,h,d,m]  (V projection, transposed)
// MODE 2: out f32 + bias[col]                 (final output projection)
// ---------------------------------------------------------------------------
template<int MODE, typename TA>
__global__ __launch_bounds__(256) void gemm_wmma_kernel(
    const TA* __restrict__ A, const float* __restrict__ B, void* __restrict__ Out,
    const float* __restrict__ bias, int M, int K, int N, float alpha)
{
    constexpr int AS = 40;                 // LDS row strides (bf16 units, 16B aligned)
    constexpr int BS = 40;
    __shared__ __align__(16) bf16_t Alds[128 * AS];
    __shared__ __align__(16) bf16_t Blds[64 * BS];   // transposed: Blds[n][k]

    const int tid  = threadIdx.x;
    const int w    = tid >> 5;
    const int lane = tid & 31;
    const int l16  = lane & 15;
    const int half = lane >> 4;
    const size_t row0 = (size_t)blockIdx.y * 128;
    const int    col0 = blockIdx.x * 64;

    v8f acc[4] = {};

    for (int k0 = 0; k0 < K; k0 += 32) {
        __syncthreads();
        // stage A tile 128x32 -> bf16 LDS
        if constexpr (sizeof(TA) == 2) {
            // bf16 input: pure copy, 512 chunks of 16B -> async-to-LDS path
            #pragma unroll
            for (int it = 0; it < 2; ++it) {
                int idx = tid + it * 256;
                int r = idx >> 2, c8 = (idx & 3) * 8;
                async_copy16(&Alds[r * AS + c8],
                             (const bf16_t*)A + (row0 + r) * (size_t)K + k0 + c8);
            }
        } else {
            // f32 input: load, convert, store (1024 chunks of 4)
            #pragma unroll
            for (int it = 0; it < 4; ++it) {
                int idx = tid + it * 256;
                int r = idx >> 3, c4 = (idx & 7) * 4;
                stage4(&Alds[r * AS + c4], A + (row0 + r) * (size_t)K + k0 + c4);
            }
        }
        // stage B tile 32x64 transposed -> Blds[n][k]
        #pragma unroll
        for (int it = 0; it < 2; ++it) {
            int idx = tid + it * 256;                   // 512 chunks of 4
            int kk = idx >> 4, n4 = (idx & 15) * 4;
            float4 t = *(const float4*)(B + (size_t)(k0 + kk) * N + col0 + n4);
            Blds[(n4 + 0) * BS + kk] = f2bf(t.x);
            Blds[(n4 + 1) * BS + kk] = f2bf(t.y);
            Blds[(n4 + 2) * BS + kk] = f2bf(t.z);
            Blds[(n4 + 3) * BS + kk] = f2bf(t.w);
        }
        wait_async();
        __syncthreads();

        // A fragment: lane holds row (w*16+l16), k = 16*half + 0..15 contiguous
        const bf16_t* ar = &Alds[(w * 16 + l16) * AS];
        v16bf a = mk16(ar + 16 * half, ar + 16 * half + 8);
        #pragma unroll
        for (int nt = 0; nt < 4; ++nt) {
            // B fragment: lane holds col (nt*16+l16), k = 16*half + 0..15 contiguous
            const bf16_t* br = &Blds[(nt * 16 + l16) * BS + 16 * half];
            v16bf b = mk16(br, br + 8);
            acc[nt] = wmma_bf16(a, b, acc[nt]);
        }
    }

    #pragma unroll
    for (int nt = 0; nt < 4; ++nt) {
        #pragma unroll
        for (int r = 0; r < 8; ++r) {
            size_t gr = row0 + (size_t)w * 16 + r + 8 * half;  // C layout: row r+8*half
            int    gc = col0 + nt * 16 + l16;                  // col = lane%16
            float  v  = acc[nt][r];
            if constexpr (MODE == 0) {
                ((bf16_t*)Out)[gr * (size_t)N + gc] = f2bf(v * alpha);
            } else if constexpr (MODE == 1) {
                int b_ = (int)(gr >> 12), m_ = (int)(gr & 4095);
                int h_ = gc >> 6,        d_ = gc & 63;
                ((bf16_t*)Out)[(((size_t)(b_ * 8 + h_) * 64 + d_) << 12) + m_] = f2bf(v * alpha);
            } else {
                ((float*)Out)[gr * (size_t)N + gc] = v + bias[gc];
            }
        }
    }
}

// ---------------------------------------------------------------------------
// Flash attention: grid (n/128, b*h), block 256 (8 waves), wave = 16 query rows.
// Q,K row-major bf16 [b*4096, 512]; Vt bf16 [b*h, 64, 4096]; O bf16 [b*4096, 512].
// Double-buffered K/V staging via async-to-LDS; per 32-key step:
// 4 WMMAs (S = Q K^T) + online softmax + 4 WMMAs (O += P V).
// ---------------------------------------------------------------------------
__global__ __launch_bounds__(256) void flash_attn_kernel(
    const bf16_t* __restrict__ Q, const bf16_t* __restrict__ K,
    const bf16_t* __restrict__ Vt, bf16_t* __restrict__ O)
{
    constexpr int KS = 72;   // Klds[key(32)][d(64)] row stride
    constexpr int VS = 40;   // Vlds[d(64)][key(32)] row stride
    constexpr int PS = 40;   // per-wave P patch [16][32] row stride
    __shared__ __align__(16) bf16_t Klds[2][32 * KS];
    __shared__ __align__(16) bf16_t Vlds[2][64 * VS];
    __shared__ __align__(16) bf16_t Plds[8 * 16 * PS];

    const int tid  = threadIdx.x;
    const int w    = tid >> 5;
    const int lane = tid & 31;
    const int l16  = lane & 15;
    const int half = lane >> 4;
    const int bh   = blockIdx.y, b = bh >> 3, h = bh & 7;
    const int qbase = blockIdx.x * 128 + w * 16;

    const bf16_t* Qg = Q + ((size_t)b * 4096) * 512 + h * 64;
    const bf16_t* Kg = K + ((size_t)b * 4096) * 512 + h * 64;
    const bf16_t* Vg = Vt + (size_t)bh * 64 * 4096;

    // Q A-fragments (d = 0..31 and 32..63) held in registers for the whole loop
    const bf16_t* qr = Qg + (size_t)(qbase + l16) * 512;
    v16bf aq0 = mk16(qr + 16 * half,      qr + 16 * half + 8);
    v16bf aq1 = mk16(qr + 32 + 16 * half, qr + 40 + 16 * half);

    v8f   acc[4] = {};
    float m_r[8], l_r[8];
    #pragma unroll
    for (int r = 0; r < 8; ++r) { m_r[r] = -1e30f; l_r[r] = 0.0f; }

    const int krow = tid >> 3, kch = tid & 7;   // K staging: 32 rows x 8 chunks
    const int vrow = tid >> 2, vch = tid & 3;   // V staging: 64 rows x 4 chunks

    auto stageKV = [&](int buf, int j0) {
        async_copy16(&Klds[buf][krow * KS + kch * 8],
                     Kg + (size_t)(j0 + krow) * 512 + kch * 8);
        async_copy16(&Vlds[buf][vrow * VS + vch * 8],
                     Vg + (size_t)vrow * 4096 + j0 + vch * 8);
    };

    stageKV(0, 0);
    wait_async();
    __syncthreads();

    int cur = 0;
    for (int j0 = 0; j0 < 4096; j0 += 32) {
        const int nxt = cur ^ 1;
        if (j0 + 32 < 4096) stageKV(nxt, j0 + 32);   // overlap next tile with compute

        // S = Q K^T for two 16-key sub-tiles (f32 accumulate)
        v8f s0 = {}, s1 = {};
        {
            const bf16_t* kr = &Klds[cur][l16 * KS];
            s0 = wmma_bf16(aq0, mk16(kr + 16 * half,      kr + 16 * half + 8), s0);
            s0 = wmma_bf16(aq1, mk16(kr + 32 + 16 * half, kr + 40 + 16 * half), s0);
            kr = &Klds[cur][(16 + l16) * KS];
            s1 = wmma_bf16(aq0, mk16(kr + 16 * half,      kr + 16 * half + 8), s1);
            s1 = wmma_bf16(aq1, mk16(kr + 32 + 16 * half, kr + 40 + 16 * half), s1);
        }

        // online softmax: row r of C-layout lives on a 16-lane half-group
        float alph[8];
        #pragma unroll
        for (int r = 0; r < 8; ++r) {
            float mx = fmaxf(s0[r], s1[r]);
            mx = fmaxf(mx, __shfl_xor(mx, 8, 32));
            mx = fmaxf(mx, __shfl_xor(mx, 4, 32));
            mx = fmaxf(mx, __shfl_xor(mx, 2, 32));
            mx = fmaxf(mx, __shfl_xor(mx, 1, 32));
            float nm = fmaxf(m_r[r], mx);
            float al = __expf(m_r[r] - nm);
            float p0 = __expf(s0[r] - nm);
            float p1 = __expf(s1[r] - nm);
            s0[r] = p0; s1[r] = p1;
            float rs = p0 + p1;
            rs += __shfl_xor(rs, 8, 32);
            rs += __shfl_xor(rs, 4, 32);
            rs += __shfl_xor(rs, 2, 32);
            rs += __shfl_xor(rs, 1, 32);
            m_r[r] = nm;
            l_r[r] = l_r[r] * al + rs;
            alph[r] = al;
        }
        #pragma unroll
        for (int nt = 0; nt < 4; ++nt)
            #pragma unroll
            for (int r = 0; r < 8; ++r)
                acc[nt][r] *= alph[r];

        // P: C-layout -> A-layout via per-wave LDS patch (same-wave DS is in-order)
        bf16_t* pw = &Plds[w * 16 * PS];
        #pragma unroll
        for (int r = 0; r < 8; ++r) {
            pw[(r + 8 * half) * PS + l16]      = f2bf(s0[r]);
            pw[(r + 8 * half) * PS + 16 + l16] = f2bf(s1[r]);
        }
        const bf16_t* pr = &pw[l16 * PS];
        v16bf aP = mk16(pr + 16 * half, pr + 16 * half + 8);

        // O += P * V  (B-frag: lane = d column, 16 consecutive keys from Vt)
        #pragma unroll
        for (int nt = 0; nt < 4; ++nt) {
            const bf16_t* vr = &Vlds[cur][(nt * 16 + l16) * VS + 16 * half];
            acc[nt] = wmma_bf16(aP, mk16(vr, vr + 8), acc[nt]);
        }

        wait_async();          // next tile landed in LDS
        __syncthreads();       // everyone done with cur, next buffer visible
        cur = nxt;
    }

    bf16_t* Og = O + ((size_t)b * 4096 + qbase) * 512 + h * 64;
    #pragma unroll
    for (int nt = 0; nt < 4; ++nt)
        #pragma unroll
        for (int r = 0; r < 8; ++r)
            Og[(size_t)(r + 8 * half) * 512 + nt * 16 + l16] = f2bf(acc[nt][r] / l_r[r]);
}

// ---------------------------------------------------------------------------
extern "C" void kernel_launch(void* const* d_in, const int* in_sizes, int n_in,
                              void* d_out, int out_size, void* d_ws, size_t ws_size,
                              hipStream_t stream) {
    (void)in_sizes; (void)n_in; (void)out_size; (void)ws_size;
    const float* x   = (const float*)d_in[0];
    const float* ctx = (const float*)d_in[1];
    const float* Wq  = (const float*)d_in[2];
    const float* Wk  = (const float*)d_in[3];
    const float* Wv  = (const float*)d_in[4];
    const float* Wo  = (const float*)d_in[5];
    const float* bo  = (const float*)d_in[6];

    const size_t NTOK = (size_t)2 * 4096;      // 8192 tokens
    const size_t INNER = 512;
    bf16_t* Qb  = (bf16_t*)d_ws;               // 8 MB each
    bf16_t* Kb  = Qb  + NTOK * INNER;
    bf16_t* Vtb = Kb  + NTOK * INNER;          // [b*h, 64, 4096]
    bf16_t* Ob  = Vtb + NTOK * INNER;

    dim3 blk(256);
    const float scale = 0.125f;                // 64^-0.5, folded into Q
    gemm_wmma_kernel<0, float ><<<dim3( 8, 64), blk, 0, stream>>>(x,   Wq, Qb,    nullptr, 8192, 1024, 512,  scale);
    gemm_wmma_kernel<0, float ><<<dim3( 8, 64), blk, 0, stream>>>(ctx, Wk, Kb,    nullptr, 8192,  768, 512,  1.0f);
    gemm_wmma_kernel<1, float ><<<dim3( 8, 64), blk, 0, stream>>>(ctx, Wv, Vtb,   nullptr, 8192,  768, 512,  1.0f);
    flash_attn_kernel<<<dim3(32, 16), blk, 0, stream>>>(Qb, Kb, Vtb, Ob);
    gemm_wmma_kernel<2, bf16_t><<<dim3(16, 64), blk, 0, stream>>>(Ob,  Wo, d_out, bo,      8192,  512, 1024, 1.0f);
}
